// SelfAttentionHead_70824010711223
// MI455X (gfx1250) — matrix operand order repaired and checked by reading, not compiled, not run on hardware
//
#include <hip/hip_runtime.h>
#include <math.h>

#define BATCH 8
#define TT    2048
#define CC    1024
#define HH    64

typedef __attribute__((ext_vector_type(16))) _Float16 v16h;
typedef __attribute__((ext_vector_type(8)))  _Float16 v8h;
typedef __attribute__((ext_vector_type(8)))  float    v8f;
typedef __attribute__((ext_vector_type(4)))  float    v4f;
typedef __attribute__((ext_vector_type(4)))  int      v4i;

static __device__ __forceinline__ v8f wmma_f16(v16h a, v16h b, v8f c) {
    // 8 args: (neg_a, A, neg_b, B, c_mod, C, reuse_a, reuse_b)
    return __builtin_amdgcn_wmma_f32_16x16x32_f16(false, a, false, b, (short)0, c, false, false);
}

// 16-byte async copy global -> LDS (ASYNCcnt-tracked, per-lane addresses)
static __device__ __forceinline__ void async_copy16(const void* g, void* l) {
#if __has_builtin(__builtin_amdgcn_global_load_async_to_lds_b128)
    __builtin_amdgcn_global_load_async_to_lds_b128(
        (__attribute__((address_space(1))) v4i*)(void*)g,
        (__attribute__((address_space(3))) v4i*)l,
        0, 0);
#else
    unsigned laddr = (unsigned)(uintptr_t)(__attribute__((address_space(3))) void*)l;
    asm volatile("global_load_async_to_lds_b128 %0, %1, off"
                 :: "v"(laddr), "v"(g) : "memory");
#endif
}

template <int N>
static __device__ __forceinline__ void wait_async() {
#if __has_builtin(__builtin_amdgcn_s_wait_asynccnt)
    __builtin_amdgcn_s_wait_asynccnt(N);
#else
    asm volatile("s_wait_asynccnt %0" :: "i"(N) : "memory");
#endif
}

// ---------------------------------------------------------------------------
// Kernel 0: transpose weights [C,H] f32 -> [H,C] f16; fold 1/sqrt(H) into WqT.
// ---------------------------------------------------------------------------
__global__ void prep_weights(const float* __restrict__ Wq,
                             const float* __restrict__ Wk,
                             const float* __restrict__ Wv,
                             _Float16* __restrict__ WqT,
                             _Float16* __restrict__ WkT,
                             _Float16* __restrict__ WvT) {
    int idx   = blockIdx.x * blockDim.x + threadIdx.x;     // 0 .. 3*C*H-1
    int which = idx / (CC * HH);
    int rem   = idx % (CC * HH);
    int c = rem / HH, h = rem % HH;
    const float* W = (which == 0) ? Wq : (which == 1) ? Wk : Wv;
    float v = W[c * HH + h];
    if (which == 0) v *= 0.125f;                            // scale = HEAD^-0.5
    _Float16* WT = (which == 0) ? WqT : (which == 1) ? WkT : WvT;
    WT[h * CC + c] = (_Float16)v;
}

// ---------------------------------------------------------------------------
// Kernel 1: QKV projection. One wave -> 16 rows x 64 cols for q, k, v.
// q,k stored row-major [B*T, H] f16; v stored transposed [B, H, T] f16.
// ---------------------------------------------------------------------------
__global__ void __launch_bounds__(256) proj_qkv(
        const float*    __restrict__ x,
        const _Float16* __restrict__ WqT,
        const _Float16* __restrict__ WkT,
        const _Float16* __restrict__ WvT,
        _Float16* __restrict__ qh,
        _Float16* __restrict__ kh,
        _Float16* __restrict__ vT) {
    const int lane = threadIdx.x & 31;
    const int wave = blockIdx.x * (blockDim.x >> 5) + (threadIdx.x >> 5);
    const int half = lane >> 4, nl = lane & 15;
    const int m0   = wave * 16;

    v8f accQ[4], accK[4], accV[4];
#pragma unroll
    for (int g = 0; g < 4; ++g)
#pragma unroll
        for (int r = 0; r < 8; ++r) { accQ[g][r] = 0.f; accK[g][r] = 0.f; accV[g][r] = 0.f; }

    const int row = m0 + nl;                 // A-layout: row = lane % 16
    const float* xr = x + (size_t)row * CC;

    for (int k0 = 0; k0 < CC; k0 += 32) {
        // ---- A fragment (x tile 16x32, fp32 -> f16) in contiguous k order (k = 16*half + e)
        const int b1 = k0 + 16 * half;       // lanes 0-15: K {0..7},  lanes 16-31: K {16..23}
        const int b2 = k0 + 16 * half + 8;   // lanes 0-15: K {8..15}, lanes 16-31: K {24..31}
        alignas(16) float f[16];
        *(v4f*)(f + 0)  = *(const v4f*)(xr + b1);
        *(v4f*)(f + 4)  = *(const v4f*)(xr + b1 + 4);
        *(v4f*)(f + 8)  = *(const v4f*)(xr + b2);
        *(v4f*)(f + 12) = *(const v4f*)(xr + b2 + 4);
        v16h a;
#pragma unroll
        for (int e = 0; e < 16; ++e) a[e] = (_Float16)f[e];

        // ---- B fragments: contiguous 32B per lane from transposed weights
        const int boff = k0 + 16 * half;     // contraction pairs per VGPR
#pragma unroll
        for (int g = 0; g < 4; ++g) {
            v16h bq = *(const v16h*)(WqT + (size_t)(16 * g + nl) * CC + boff);
            accQ[g] = wmma_f16(a, bq, accQ[g]);
            v16h bk = *(const v16h*)(WkT + (size_t)(16 * g + nl) * CC + boff);
            accK[g] = wmma_f16(a, bk, accK[g]);
            v16h bv = *(const v16h*)(WvT + (size_t)(16 * g + nl) * CC + boff);
            accV[g] = wmma_f16(a, bv, accV[g]);
        }
    }

    // ---- stores (C/D layout: row = r + 8*half, col = 16*g + nl)
    const int b  = m0 / TT;
    const int t0 = m0 % TT;
#pragma unroll
    for (int g = 0; g < 4; ++g) {
#pragma unroll
        for (int r = 0; r < 8; ++r) {
            const int trow = m0 + r + 8 * half;
            const int h    = 16 * g + nl;
            qh[(size_t)trow * HH + h] = (_Float16)accQ[g][r];
            kh[(size_t)trow * HH + h] = (_Float16)accK[g][r];
            vT[((size_t)b * HH + h) * TT + (t0 + r + 8 * half)] = (_Float16)accV[g][r];
        }
    }
}

// ---------------------------------------------------------------------------
// Kernel 2: causal flash attention.
// Block = one batch x 8 consecutive 16-query tiles (wave w -> qtile qg*8+w).
// K/V tiles staged per block in double-buffered LDS via
// GLOBAL_LOAD_ASYNC_TO_LDS_B128, software-pipelined so tile i+1's async DMA
// overlaps tile i's WMMA work; uniform key loop -> in-loop barriers are legal.
// ---------------------------------------------------------------------------
__global__ void __launch_bounds__(256) attn(
        const _Float16* __restrict__ qh,
        const _Float16* __restrict__ kh,
        const _Float16* __restrict__ vT,
        float* __restrict__ out) {
    __shared__ __align__(32) _Float16 kTile[2][32 * 64];   // [key][h]   2 x 4 KB
    __shared__ __align__(32) _Float16 vTile[2][64 * 32];   // [h][key]   2 x 4 KB
    __shared__ __align__(32) _Float16 pbuf[8 * 16 * 32];   // per-wave P scratch

    const int lane = threadIdx.x & 31;
    const int wid  = threadIdx.x >> 5;
    const int half = lane >> 4, nl = lane & 15;
    const int b    = blockIdx.x >> 4;        // batch
    const int qg   = blockIdx.x & 15;        // q-tile group (8 tiles)
    const int q0   = (qg * 8 + wid) * 16;    // this wave's query tile

    const _Float16* qb  = qh + (size_t)b * TT * HH;
    const _Float16* kb_ = kh + (size_t)b * TT * HH;
    const _Float16* vb  = vT + (size_t)b * HH * TT;
    _Float16* pl = pbuf + wid * 16 * 32;

    const int idx = wid * 32 + lane;         // 0..255: this thread's copy slot
    const int vh  = idx >> 2, vp = idx & 3;  // V tile: row h, 16B part

    // issue this thread's 2 async 16B transfers for key block kb into buffer ib
    auto stage = [&](int kb, int ib) {
        async_copy16((const char*)kb_ + ((size_t)kb * HH) * 2 + idx * 16,
                     (char*)kTile[ib] + idx * 16);
        async_copy16((const char*)vb + ((size_t)vh * TT + kb) * 2 + vp * 16,
                     (char*)vTile[ib] + vh * 64 + vp * 16);
    };

    // ---- Q A-fragments (2 fragments cover contraction h=0..63)
    v16h qf[2];
#pragma unroll
    for (int j = 0; j < 2; ++j) {
        const _Float16* qrow = qb + (size_t)(q0 + nl) * HH;
        v8h lo = *(const v8h*)(qrow + 32 * j + 16 * half);
        v8h hi = *(const v8h*)(qrow + 32 * j + 16 * half + 8);
#pragma unroll
        for (int e = 0; e < 8; ++e) { qf[j][e] = lo[e]; qf[j][8 + e] = hi[e]; }
    }

    v8f o[4];
    float m[8], l[8];
#pragma unroll
    for (int g = 0; g < 4; ++g)
#pragma unroll
        for (int r = 0; r < 8; ++r) o[g][r] = 0.f;
#pragma unroll
    for (int r = 0; r < 8; ++r) { m[r] = -INFINITY; l[r] = 0.f; }

    const int kend_own = q0 + 16;            // this wave's causal bound
    const int kend_max = (qg * 8 + 8) * 16;  // block-uniform bound (>= 128)

    stage(0, 0);                             // prologue: tile 0 in flight

    int ibuf = 0;
    for (int kb = 0; kb < kend_max; kb += 32, ibuf ^= 1) {
        // pipeline: issue next tile's DMA before waiting on the current one
        const bool have_next = (kb + 32) < kend_max;
        if (have_next) {
            stage(kb + 32, ibuf ^ 1);
            wait_async<2>();                 // in-order: <=2 outstanding => tile kb done
        } else {
            wait_async<0>();
        }
        __syncthreads();                     // current tile visible to all waves

        if (kb < kend_own) {
            const _Float16* kT = kTile[ibuf];
            const _Float16* vTl = vTile[ibuf];

            // ---- S = Q K^T (two 16-key subtiles, contraction 64 = 2 WMMAs)
            v8f S[2];
#pragma unroll
            for (int s = 0; s < 2; ++s) {
                const _Float16* krow = kT + (16 * s + nl) * HH;
                v16h kf0 = *(const v16h*)(krow + 16 * half);
                v16h kf1 = *(const v16h*)(krow + 32 + 16 * half);
                v8f sacc;
#pragma unroll
                for (int r = 0; r < 8; ++r) sacc[r] = 0.f;
                sacc = wmma_f16(qf[0], kf0, sacc);
                sacc = wmma_f16(qf[1], kf1, sacc);
                if (kb + 16 * s + 15 > q0) {          // touches/crosses diagonal
                    const int key = kb + 16 * s + nl;
#pragma unroll
                    for (int r = 0; r < 8; ++r) {
                        const int t = q0 + r + 8 * half;
                        if (key > t) sacc[r] = -1.0e30f;
                    }
                }
                S[s] = sacc;
            }

            // ---- online softmax (rows at fixed vgpr across each 16-lane group)
            float alpha[8];
#pragma unroll
            for (int r = 0; r < 8; ++r) {
                float rm = fmaxf(S[0][r], S[1][r]);
                rm = fmaxf(rm, __shfl_xor(rm, 1, 32));
                rm = fmaxf(rm, __shfl_xor(rm, 2, 32));
                rm = fmaxf(rm, __shfl_xor(rm, 4, 32));
                rm = fmaxf(rm, __shfl_xor(rm, 8, 32));
                const float mnew = fmaxf(m[r], rm);
                alpha[r] = __expf(m[r] - mnew);
                m[r] = mnew;
            }
#pragma unroll
            for (int s = 0; s < 2; ++s)
#pragma unroll
                for (int r = 0; r < 8; ++r) S[s][r] = __expf(S[s][r] - m[r]);
#pragma unroll
            for (int r = 0; r < 8; ++r) {
                float rs = S[0][r] + S[1][r];
                rs += __shfl_xor(rs, 1, 32);
                rs += __shfl_xor(rs, 2, 32);
                rs += __shfl_xor(rs, 4, 32);
                rs += __shfl_xor(rs, 8, 32);
                l[r] = l[r] * alpha[r] + rs;
            }
#pragma unroll
            for (int g = 0; g < 4; ++g)
#pragma unroll
                for (int r = 0; r < 8; ++r) o[g][r] *= alpha[r];

            // ---- P: C-layout -> A-layout via per-wave LDS scratch
#pragma unroll
            for (int s = 0; s < 2; ++s)
#pragma unroll
                for (int r = 0; r < 8; ++r)
                    pl[(r + 8 * half) * 32 + 16 * s + nl] = (_Float16)S[s][r];

            asm volatile("s_wait_dscnt 0" ::: "memory");   // same-wave LDS RAW

            v16h pf;
            {
                const _Float16* prow = pl + nl * 32;       // A row = lane % 16
                v8h lo = *(const v8h*)(prow + 16 * half);
                v8h hi = *(const v8h*)(prow + 16 * half + 8);
#pragma unroll
                for (int e = 0; e < 8; ++e) { pf[e] = lo[e]; pf[8 + e] = hi[e]; }
            }

            // ---- O += P V : B-fragments contiguous from LDS V tile
#pragma unroll
            for (int g = 0; g < 4; ++g) {
                v16h vf = *(const v16h*)(vTl + (16 * g + nl) * 32 + 16 * half);
                o[g] = wmma_f16(pf, vf, o[g]);
            }
        }

        __syncthreads();                     // all reads done before buffer reuse
    }

    // ---- epilogue: normalize by row sum, fp32 out [B,T,H]
    float* ob = out + ((size_t)b * TT + q0) * HH;
#pragma unroll
    for (int r = 0; r < 8; ++r) {
        const float inv = 1.0f / l[r];
#pragma unroll
        for (int g = 0; g < 4; ++g)
            ob[(size_t)(r + 8 * half) * HH + 16 * g + nl] = o[g][r] * inv;
    }
}

// ---------------------------------------------------------------------------
extern "C" void kernel_launch(void* const* d_in, const int* in_sizes, int n_in,
                              void* d_out, int out_size, void* d_ws, size_t ws_size,
                              hipStream_t stream) {
    const float* x  = (const float*)d_in[0];
    const float* Wq = (const float*)d_in[1];
    const float* Wk = (const float*)d_in[2];
    const float* Wv = (const float*)d_in[3];

    char* ws = (char*)d_ws;
    const size_t MB = 1024ull * 1024ull;
    _Float16* qh  = (_Float16*)(ws);                  // [B*T, H] f16   (2 MB)
    _Float16* kh  = (_Float16*)(ws + 2 * MB);         // [B*T, H] f16   (2 MB)
    _Float16* vT  = (_Float16*)(ws + 4 * MB);         // [B, H, T] f16  (2 MB)
    _Float16* WqT = (_Float16*)(ws + 6 * MB);         // [H, C] f16 (scaled)
    _Float16* WkT = WqT + CC * HH;
    _Float16* WvT = WkT + CC * HH;

    prep_weights<<<(3 * CC * HH) / 256, 256, 0, stream>>>(Wq, Wk, Wv, WqT, WkT, WvT);
    proj_qkv<<<(BATCH * TT / 16) / 8, 256, 0, stream>>>(x, WqT, WkT, WvT, qh, kh, vT);
    attn<<<BATCH * 16, 256, 0, stream>>>(qh, kh, vT, (float*)d_out);
}
